// Seq2seq_Model_3195455668451
// MI455X (gfx1250) — compile-verified, matrix operand order checked by reading, not run on hardware
//
#include <hip/hip_runtime.h>
#include <cstdint>

// ---------------------------------------------------------------- constants
constexpr int EMB = 256, HID = 512, VOC2 = 32002;
constexpr int B = 16, S = 128, TD = 128;
constexpr int DHID = 2 * HID;          // 1024
constexpr int MROWS = B * S;           // 2048 (== B*TD)

// ---------------------------------------------------------------- vec types
typedef __attribute__((ext_vector_type(16))) __bf16 v16bf;
typedef __attribute__((ext_vector_type(8)))  __bf16 v8bf;
typedef __attribute__((ext_vector_type(8)))  float  v8f;

__device__ __forceinline__ float sigmoidf_(float x) {
    return 1.0f / (1.0f + __expf(-x));
}

// generic LDS pointer -> 32-bit LDS byte offset (aperture low bits)
__device__ __forceinline__ uint32_t lds_off(const void* p) {
    return (uint32_t)(uintptr_t)p;
}

// CDNA5 async DMA: LDS[vdst] = MEM[vaddr], 16B per lane, ASYNCcnt-tracked
__device__ __forceinline__ void async_b128(uint32_t ldsoff, const void* g) {
    asm volatile("global_load_async_to_lds_b128 %0, %1, off"
                 :: "v"(ldsoff), "v"((uint64_t)(uintptr_t)g)
                 : "memory");
}

template<int N>
__device__ __forceinline__ void wait_async() {
#if __has_builtin(__builtin_amdgcn_s_wait_asynccnt)
    __builtin_amdgcn_s_wait_asynccnt((short)N);
#else
    asm volatile("s_wait_asynccnt %0" :: "i"(N) : "memory");
#endif
}

// =====================================================================
// WMMA GEMM:  C[M,N] = A[M,K] * op(B) + bias    (A,B bf16; acc fp32)
//   TRANSB=false : B is [N,K] row-major (weight form, C = A * B^T)
//   TRANSB=true  : B is [K,N] row-major (C = A * B)
//   ASYNC        : double-buffered global_load_async_to_lds_b128 staging
//                  (requires BM==128, TRANSB==false, 16B-aligned rows)
// Requirements: M % BM == 0, K % 32 == 0, lda/ldb % 8 == 0.
// N ragged: B source rows clamped (garbage only feeds unstored C cols),
// C stores guarded. blockIdx = (Ntile, Mtile, batch).
// =====================================================================
template<int BM, int BN, int TM, int TN, bool TRANSB, bool ASYNC, typename CT>
__global__ __launch_bounds__(256) void gemm_wmma(
    const __bf16* __restrict__ A, int lda, long long sA,
    const __bf16* __restrict__ Bm, int ldb, long long sB,
    CT* __restrict__ C, int ldc, long long sC,
    const float* __restrict__ bias,
    int M, int N, int K)
{
    constexpr int BK = 32;
    constexpr int NBUF = ASYNC ? 2 : 1;
    __shared__ alignas(16) __bf16 ldsA[NBUF][BM * BK];
    __shared__ alignas(16) __bf16 ldsB[NBUF][BN * BK];

    A  += (long long)blockIdx.z * sA;
    Bm += (long long)blockIdx.z * sB;
    C  += (long long)blockIdx.z * sC;

    const int n0   = blockIdx.x * BN;
    const int m0   = blockIdx.y * BM;
    const int tid  = threadIdx.x;
    const int wid  = tid >> 5;
    const int lane = tid & 31;
    const int lrow  = lane & 15;
    const int lhalf = lane >> 4;

    constexpr int WN = BN / (16 * TN);     // waves along N
    const int wn = wid % WN;
    const int wm = wid / WN;

    v8f acc[TM][TN];
    #pragma unroll
    for (int mi = 0; mi < TM; ++mi)
        #pragma unroll
        for (int ni = 0; ni < TN; ++ni) {
            v8f z = {};
            acc[mi][ni] = z;
        }

    // ---- fragment compute from LDS buffer cb
    auto compute = [&](int cb) {
        v16bf afr[TM], bfr[TN];
        #pragma unroll
        for (int mi = 0; mi < TM; ++mi) {
            int r = wm * TM * 16 + mi * 16 + lrow;
            const __bf16* p = &ldsA[cb][r * BK + 8 * lhalf];
            union { v16bf v; v8bf h[2]; } u;
            u.h[0] = *(const v8bf*)p;
            u.h[1] = *(const v8bf*)(p + 16);
            afr[mi] = u.v;
        }
        #pragma unroll
        for (int ni = 0; ni < TN; ++ni) {
            int r = wn * TN * 16 + ni * 16 + lrow;
            const __bf16* p = &ldsB[cb][r * BK + 8 * lhalf];
            union { v16bf v; v8bf h[2]; } u;
            u.h[0] = *(const v8bf*)p;
            u.h[1] = *(const v8bf*)(p + 16);
            bfr[ni] = u.v;
        }
        #pragma unroll
        for (int mi = 0; mi < TM; ++mi)
            #pragma unroll
            for (int ni = 0; ni < TN; ++ni)
                acc[mi][ni] = __builtin_amdgcn_wmma_f32_16x16x32_bf16(
                    false, afr[mi], false, bfr[ni],
                    (short)0, acc[mi][ni], false, false);
    };

    const int nk = K / BK;

    if constexpr (ASYNC && !TRANSB) {
        // 16B chunks: A tile BM*4 chunks, B tile BN*4 chunks; each wave
        // issues exactly 2+2 async ops per tile (uniform => imm wait legal).
        auto stage = [&](int buf, int kk) {
            #pragma unroll
            for (int c = tid; c < BM * 4; c += 256) {
                int m = c >> 2, ke = (c & 3) * 8;
                async_b128(lds_off(&ldsA[buf][0]) + c * 16,
                           A + (long long)(m0 + m) * lda + kk + ke);
            }
            #pragma unroll
            for (int c = tid; c < BN * 4; c += 256) {
                int n = c >> 2, ke = (c & 3) * 8;
                int nc = n0 + n; if (nc >= N) nc = N - 1;   // clamp (see hdr)
                async_b128(lds_off(&ldsB[buf][0]) + c * 16,
                           Bm + (long long)nc * ldb + kk + ke);
            }
        };
        stage(0, 0);
        for (int i = 0; i < nk; ++i) {
            int cb = i & 1;
            if (i + 1 < nk) { stage(cb ^ 1, (i + 1) * BK); wait_async<4>(); }
            else           { wait_async<0>(); }
            __syncthreads();
            compute(cb);
            __syncthreads();
        }
    } else {
        for (int i = 0; i < nk; ++i) {
            int kk = i * BK;
            __syncthreads();
            for (int idx = tid; idx < BM * BK; idx += 256) {
                int m = idx / BK, k = idx % BK;
                ldsA[0][idx] = A[(long long)(m0 + m) * lda + (kk + k)];
            }
            if constexpr (!TRANSB) {
                for (int idx = tid; idx < BN * BK; idx += 256) {
                    int n = idx / BK, k = idx % BK;
                    __bf16 v = (__bf16)0.0f;
                    if (n0 + n < N) v = Bm[(long long)(n0 + n) * ldb + (kk + k)];
                    ldsB[0][idx] = v;
                }
            } else {
                for (int idx = tid; idx < BN * BK; idx += 256) {
                    int k = idx / BN, n = idx % BN;     // coalesced over n
                    __bf16 v = (__bf16)0.0f;
                    if (n0 + n < N) v = Bm[(long long)(kk + k) * ldb + (n0 + n)];
                    ldsB[0][n * BK + k] = v;
                }
            }
            __syncthreads();
            compute(0);
        }
    }

    // ---- epilogue: C layout VGPR i -> M = i + 8*lhalf, N = lrow
    #pragma unroll
    for (int mi = 0; mi < TM; ++mi)
        #pragma unroll
        for (int ni = 0; ni < TN; ++ni) {
            int cbase = n0 + wn * TN * 16 + ni * 16 + lrow;
            if (cbase < N) {
                float bv = bias ? bias[cbase] : 0.0f;
                #pragma unroll
                for (int i = 0; i < 8; ++i) {
                    int r = m0 + wm * TM * 16 + mi * 16 + lhalf * 8 + i;
                    C[(long long)r * ldc + cbase] = (CT)(acc[mi][ni][i] + bv);
                }
            }
        }
}

// =====================================================================
// Elementwise / utility kernels
// =====================================================================
__global__ void f2bf_kernel(const float* __restrict__ s,
                            __bf16* __restrict__ d, long long n)
{
    long long i = (long long)blockIdx.x * 256 + threadIdx.x;
    if (i < n) d[i] = (__bf16)s[i];
}

__global__ void embed_bf_kernel(const int* __restrict__ tok,
                                const float* __restrict__ emb,
                                __bf16* __restrict__ out, int ntok)
{
    long long i = (long long)blockIdx.x * 256 + threadIdx.x;
    if (i >= (long long)ntok * EMB) return;
    int e = (int)(i % EMB);
    long long t = i / EMB;
    out[i] = (__bf16)emb[(long long)tok[t] * EMB + e];
}

__global__ void zero_f_kernel(float* __restrict__ p, int n)
{
    int i = blockIdx.x * 256 + threadIdx.x;
    if (i < n) p[i] = 0.0f;
}

__global__ void zero_bf_kernel(__bf16* __restrict__ p, int n)
{
    int i = blockIdx.x * 256 + threadIdx.x;
    if (i < n) p[i] = (__bf16)0.0f;
}

// h_dec0 = raw flatten of stacked [hf;hb] (torch .view semantics);
// hf,hb are adjacent in the workspace so this is a flat copy.
__global__ void init_hd_kernel(const float* __restrict__ hfb,
                               float* __restrict__ h_d,
                               __bf16* __restrict__ hd_bf, int n)
{
    int i = blockIdx.x * 256 + threadIdx.x;
    if (i < n) { float v = hfb[i]; h_d[i] = v; hd_bf[i] = (__bf16)v; }
}

// encoder gate fusion for one time step (both directions)
__global__ void enc_gates(const float* __restrict__ xw_f,
                          const float* __restrict__ xw_b,
                          const float* __restrict__ gh_f,
                          const float* __restrict__ gh_b,
                          float* __restrict__ h_f,
                          float* __restrict__ h_b,
                          __bf16* __restrict__ hf_bf,
                          __bf16* __restrict__ hb_bf,
                          __bf16* __restrict__ out_enc_bf, int t)
{
    int idx = blockIdx.x * 256 + threadIdx.x;      // 2*16*512 threads
    if (idx >= 2 * B * HID) return;
    int dir = idx / (B * HID);
    int r = idx % (B * HID);
    int b = r / HID, j = r % HID;
    const float* xw = dir ? xw_b : xw_f;
    const float* gh = dir ? gh_b : gh_f;
    float*  h  = dir ? h_b   : h_f;
    __bf16* hb = dir ? hb_bf : hf_bf;
    int s = dir ? (S - 1 - t) : t;

    const float* xrow = xw + ((long long)b * S + s) * (3 * HID);
    const float* grow = gh + b * (3 * HID);
    float rr = sigmoidf_(xrow[j]           + grow[j]);
    float zz = sigmoidf_(xrow[HID + j]     + grow[HID + j]);
    float nn = tanhf(xrow[2 * HID + j] + rr * grow[2 * HID + j]);
    float hp = h[b * HID + j];
    float hn = (1.0f - zz) * nn + zz * hp;
    h[b * HID + j]  = hn;
    hb[b * HID + j] = (__bf16)hn;
    out_enc_bf[((long long)b * S + s) * DHID + dir * HID + j] = (__bf16)hn;
}

// decoder gate fusion; hidden -> fp32 state + bf16 mirror + feat left half
__global__ void dec_gates(const float* __restrict__ yw,
                          const float* __restrict__ gh_d,
                          float* __restrict__ h_d,
                          __bf16* __restrict__ hd_bf,
                          __bf16* __restrict__ feat_bf, int t)
{
    int idx = blockIdx.x * 256 + threadIdx.x;      // 16*1024 threads
    if (idx >= B * DHID) return;
    int b = idx / DHID, j = idx % DHID;
    const float* xrow = yw + ((long long)b * TD + t) * (3 * DHID);
    const float* grow = gh_d + b * (3 * DHID);
    float rr = sigmoidf_(xrow[j]            + grow[j]);
    float zz = sigmoidf_(xrow[DHID + j]     + grow[DHID + j]);
    float nn = tanhf(xrow[2 * DHID + j] + rr * grow[2 * DHID + j]);
    float hp = h_d[idx];
    float hn = (1.0f - zz) * nn + zz * hp;
    h_d[idx]  = hn;
    hd_bf[idx] = (__bf16)hn;
    feat_bf[((long long)b * TD + t) * (2 * DHID) + j] = (__bf16)hn;
}

// row softmax over S=128 (one block of 128 threads per row), bf16 out
__global__ void softmax128(const float* __restrict__ scores,
                           __bf16* __restrict__ out_bf)
{
    __shared__ float red[128];
    const float* p = scores + (long long)blockIdx.x * S;
    int t = threadIdx.x;
    float v = p[t];
    red[t] = v; __syncthreads();
    for (int off = 64; off > 0; off >>= 1) {
        if (t < off) red[t] = fmaxf(red[t], red[t + off]);
        __syncthreads();
    }
    float m = red[0]; __syncthreads();
    float e = __expf(v - m);
    red[t] = e; __syncthreads();
    for (int off = 64; off > 0; off >>= 1) {
        if (t < off) red[t] += red[t + off];
        __syncthreads();
    }
    out_bf[(long long)blockIdx.x * S + t] = (__bf16)(e / red[0]);
}

// =====================================================================
// host orchestration
// =====================================================================
extern "C" void kernel_launch(void* const* d_in, const int* in_sizes, int n_in,
                              void* d_out, int out_size, void* d_ws, size_t ws_size,
                              hipStream_t stream)
{
    (void)in_sizes; (void)n_in; (void)out_size; (void)ws_size;
    const int*   inp      = (const int*)d_in[0];
    const int*   dec_inp  = (const int*)d_in[1];
    const float* emb      = (const float*)d_in[2];
    const float* eWih_f   = (const float*)d_in[3];
    const float* eWhh_f   = (const float*)d_in[4];
    const float* ebih_f   = (const float*)d_in[5];
    const float* ebhh_f   = (const float*)d_in[6];
    const float* eWih_b   = (const float*)d_in[7];
    const float* eWhh_b   = (const float*)d_in[8];
    const float* ebih_b   = (const float*)d_in[9];
    const float* ebhh_b   = (const float*)d_in[10];
    const float* dWih     = (const float*)d_in[11];
    const float* dWhh     = (const float*)d_in[12];
    const float* dbih     = (const float*)d_in[13];
    const float* dbhh     = (const float*)d_in[14];
    const float* att      = (const float*)d_in[15];
    const float* fc_W     = (const float*)d_in[16];
    const float* fc_b     = (const float*)d_in[17];
    float* out = (float*)d_out;

    // ---- workspace carve: fp32 region, then bf16 region
    float* w = (float*)d_ws;
    size_t off = 0;
    auto allocf = [&](size_t n) { float* p = w + off; off += n; return p; };
    float* xw_f    = allocf((size_t)MROWS * 3 * HID);
    float* xw_b    = allocf((size_t)MROWS * 3 * HID);
    float* yw      = allocf((size_t)MROWS * 3 * DHID);
    float* scores  = allocf((size_t)B * TD * S);
    float* gh_f    = allocf((size_t)B * 3 * HID);
    float* gh_b    = allocf((size_t)B * 3 * HID);
    float* gh_d    = allocf((size_t)B * 3 * DHID);
    float* h_f     = allocf((size_t)B * HID);      // h_f,h_b adjacent
    float* h_b     = allocf((size_t)B * HID);
    float* h_d     = allocf((size_t)B * DHID);

    __bf16* wb = (__bf16*)(w + off);
    size_t boff = 0;
    auto allocb = [&](size_t n) { __bf16* p = wb + boff; boff += n; return p; };
    __bf16* x_enc_bf   = allocb((size_t)B * S * EMB);
    __bf16* y_dec_bf   = allocb((size_t)B * TD * EMB);
    __bf16* wihf_bf    = allocb((size_t)3 * HID * EMB);
    __bf16* wihb_bf    = allocb((size_t)3 * HID * EMB);
    __bf16* wihd_bf    = allocb((size_t)3 * DHID * EMB);
    __bf16* whhf_bf    = allocb((size_t)3 * HID * HID);
    __bf16* whhb_bf    = allocb((size_t)3 * HID * HID);
    __bf16* whhd_bf    = allocb((size_t)3 * DHID * DHID);
    __bf16* att_bf     = allocb((size_t)DHID * DHID);
    __bf16* fcW_bf     = allocb((size_t)VOC2 * 2 * DHID);  // 131 MB: fits L2
    __bf16* out_enc_bf = allocb((size_t)B * S * DHID);
    __bf16* feat_bf    = allocb((size_t)MROWS * 2 * DHID);
    __bf16* proj_bf    = allocb((size_t)MROWS * DHID);
    __bf16* scores_bf  = allocb((size_t)B * TD * S);
    __bf16* hf_bf      = allocb((size_t)B * HID);
    __bf16* hb_bf      = allocb((size_t)B * HID);
    __bf16* hd_bf      = allocb((size_t)B * DHID);

    auto conv = [&](const float* s, __bf16* d, long long n) {
        f2bf_kernel<<<(unsigned)((n + 255) / 256), 256, 0, stream>>>(s, d, n);
    };

    // ---- weight conversion (once per call; amortized over the whole net)
    conv(eWih_f, wihf_bf, (long long)3 * HID * EMB);
    conv(eWih_b, wihb_bf, (long long)3 * HID * EMB);
    conv(dWih,   wihd_bf, (long long)3 * DHID * EMB);
    conv(eWhh_f, whhf_bf, (long long)3 * HID * HID);
    conv(eWhh_b, whhb_bf, (long long)3 * HID * HID);
    conv(dWhh,   whhd_bf, (long long)3 * DHID * DHID);
    conv(att,    att_bf,  (long long)DHID * DHID);
    conv(fc_W,   fcW_bf,  (long long)VOC2 * 2 * DHID);

    // ---- embeddings (gather + bf16 cast fused)
    embed_bf_kernel<<<(B * S * EMB) / 256, 256, 0, stream>>>(inp, emb, x_enc_bf, B * S);
    embed_bf_kernel<<<(B * TD * EMB) / 256, 256, 0, stream>>>(dec_inp, emb, y_dec_bf, B * TD);

    // ---- input projections (async-staged WMMA GEMMs, bias = bih)
    gemm_wmma<128,128,2,4,false,true,float><<<dim3(3*HID/128, MROWS/128, 1), 256, 0, stream>>>(
        x_enc_bf, EMB, 0, wihf_bf, EMB, 0, xw_f, 3*HID, 0, ebih_f, MROWS, 3*HID, EMB);
    gemm_wmma<128,128,2,4,false,true,float><<<dim3(3*HID/128, MROWS/128, 1), 256, 0, stream>>>(
        x_enc_bf, EMB, 0, wihb_bf, EMB, 0, xw_b, 3*HID, 0, ebih_b, MROWS, 3*HID, EMB);
    gemm_wmma<128,128,2,4,false,true,float><<<dim3(3*DHID/128, MROWS/128, 1), 256, 0, stream>>>(
        y_dec_bf, EMB, 0, wihd_bf, EMB, 0, yw, 3*DHID, 0, dbih, MROWS, 3*DHID, EMB);

    // ---- encoder recurrence
    zero_f_kernel<<<(2 * B * HID + 255) / 256, 256, 0, stream>>>(h_f, 2 * B * HID);
    zero_bf_kernel<<<(2 * B * HID + 255) / 256, 256, 0, stream>>>(hf_bf, 2 * B * HID);
    for (int t = 0; t < S; ++t) {
        gemm_wmma<16,128,1,1,false,false,float><<<dim3(3*HID/128, 1, 1), 256, 0, stream>>>(
            hf_bf, HID, 0, whhf_bf, HID, 0, gh_f, 3*HID, 0, ebhh_f, B, 3*HID, HID);
        gemm_wmma<16,128,1,1,false,false,float><<<dim3(3*HID/128, 1, 1), 256, 0, stream>>>(
            hb_bf, HID, 0, whhb_bf, HID, 0, gh_b, 3*HID, 0, ebhh_b, B, 3*HID, HID);
        enc_gates<<<(2 * B * HID) / 256, 256, 0, stream>>>(
            xw_f, xw_b, gh_f, gh_b, h_f, h_b, hf_bf, hb_bf, out_enc_bf, t);
    }

    // ---- h_dec0 (hf,hb adjacent -> flat copy, matches torch .view)
    init_hd_kernel<<<(B * DHID) / 256, 256, 0, stream>>>(h_f, h_d, hd_bf, B * DHID);

    // ---- decoder recurrence; hidden written to feat[:, :DHID]
    for (int t = 0; t < TD; ++t) {
        gemm_wmma<16,128,1,1,false,false,float><<<dim3(3*DHID/128, 1, 1), 256, 0, stream>>>(
            hd_bf, DHID, 0, whhd_bf, DHID, 0, gh_d, 3*DHID, 0, dbhh, B, 3*DHID, DHID);
        dec_gates<<<(B * DHID) / 256, 256, 0, stream>>>(yw, gh_d, h_d, hd_bf, feat_bf, t);
    }

    // ---- attention: proj = out_dec @ att   (TRANSB, bf16 C)
    gemm_wmma<128,128,2,4,true,false,__bf16><<<dim3(DHID/128, MROWS/128, 1), 256, 0, stream>>>(
        feat_bf, 2*DHID, 0, att_bf, DHID, 0, proj_bf, DHID, 0, nullptr, MROWS, DHID, DHID);

    // scores[b] = proj[b] @ out_enc[b]^T  (batched over z, async staging)
    gemm_wmma<128,128,2,4,false,true,float><<<dim3(1, 1, B), 256, 0, stream>>>(
        proj_bf, DHID, (long long)TD * DHID,
        out_enc_bf, DHID, (long long)S * DHID,
        scores, S, (long long)TD * S, nullptr, TD, S, DHID);

    softmax128<<<B * TD, 128, 0, stream>>>(scores, scores_bf);

    // att_cxt[b] = wgt[b] @ out_enc[b]  -> feat[:, DHID:]  (bf16 C)
    gemm_wmma<128,128,2,4,true,false,__bf16><<<dim3(DHID/128, 1, B), 256, 0, stream>>>(
        scores_bf, S, (long long)TD * S,
        out_enc_bf, DHID, (long long)S * DHID,
        feat_bf + DHID, 2*DHID, (long long)TD * 2 * DHID,
        nullptr, TD, DHID, S);

    // ---- final FC: out = feat @ fc_W^T + fc_b (dominant 268 GF GEMM;
    //      bf16 weights = 131 MB -> resident in 192 MB L2 across M-tiles)
    gemm_wmma<128,128,2,4,false,true,float><<<dim3((VOC2 + 127) / 128, MROWS / 128, 1), 256, 0, stream>>>(
        feat_bf, 2*DHID, 0, fcW_bf, 2*DHID, 0, out, VOC2, 0, fc_b,
        MROWS, VOC2, 2*DHID);
}
